// JointEncoder_43997644980914
// MI455X (gfx1250) — matrix operand order repaired and checked by reading, not compiled, not run on hardware
//
#include <hip/hip_runtime.h>

typedef __attribute__((ext_vector_type(16))) _Float16 v16h;
typedef __attribute__((ext_vector_type(8)))  float    v8f;

#define BB   16
#define N1   1024
#define M1C  1023
#define M2C  338
#define KNN  64
#define NEGF (-1.0e9f)

// ---------------------------------------------------------------------------
// WMMA fragment helpers (CDNA5 16x16x32 f16 layouts, CDNA5 ISA)
// ---------------------------------------------------------------------------

// A 16x32 f16 from LDS row-major [rows][stride]:
// lanes 0-15 : M=lane,    half h -> K=kBase+h
// lanes 16-31: M=lane-16, half h -> K=kBase+16+h
// One contiguous 32B run per lane -> 2x ds_load_b128.
__device__ inline v16h load_A_frag(const _Float16* lds, int rowBase, int stride, int kBase) {
  int lane = threadIdx.x & 31;
  int hi = lane >> 4;
  int m  = lane & 15;
  const _Float16* rp = lds + (size_t)(rowBase + m) * stride + kBase + hi * 16;
  v16h a;
#pragma unroll
  for (int h = 0; h < 8; ++h)  a[h] = rp[h];
#pragma unroll
  for (int h = 8; h < 16; ++h) a[h] = rp[h];
  return a;
}

// B 32x16 f16 from global TRANSPOSED weights WT[Nc][Kpad]:
// lanes 0-15 : N=nBase+lane,    half h -> K=kBase+h
// lanes 16-31: N=nBase+lane-16, half h -> K=kBase+16+h
// Per lane: 16 contiguous halves (32B) -> 2x global_load_b128 (L2-resident).
__device__ inline v16h load_B_frag(const _Float16* WT, int Kpad, int kBase, int nBase) {
  int lane = threadIdx.x & 31;
  int hi = lane >> 4;
  const _Float16* p = WT + (size_t)(nBase + (lane & 15)) * Kpad + kBase + hi * 16;
  v16h b;
#pragma unroll
  for (int h = 0; h < 16; ++h) b[h] = p[h];
  return b;
}

// C/D 16x16 f32: VGPR r -> M=r (lanes 0-15) or M=r+8 (lanes 16-31), N=lane&15.
__device__ inline void store_C_relu(v8f c, _Float16* lds, int rowBase, int stride,
                                    int nBase, const float* bias) {
  int lane = threadIdx.x & 31;
  int hi = lane >> 4;
  int n  = lane & 15;
  float bv = bias[nBase + n];
  _Float16* p = lds + nBase + n;
#pragma unroll
  for (int r = 0; r < 8; ++r) {
    float v = c[r] + bv;
    v = v > 0.f ? v : 0.f;
    p[(size_t)(rowBase + r + hi * 8) * stride] = (_Float16)v;
  }
}

// One Linear+ReLU layer: LDS f16 [ROWS][Kin] -> LDS f16 [ROWS][Nc].
// Kin must be a multiple of 32, Nc/ROWS multiples of 16. Whole block participates.
// WT is transposed f16 weights [Nc][Kin].
__device__ inline void mlp_layer(const _Float16* __restrict__ in, _Float16* __restrict__ out,
                                 const _Float16* __restrict__ WT, const float* __restrict__ bias,
                                 int ROWS, int Kin, int Nc, int wave, int nwaves) {
  int NT = Nc >> 4, MT = ROWS >> 4, KT = Kin >> 5;
  for (int tix = wave; tix < NT * MT; tix += nwaves) {
    int nt = tix / MT, mt = tix - nt * MT;
    v8f c = {};
    for (int kt = 0; kt < KT; ++kt) {
      v16h a  = load_A_frag(in, mt << 4, Kin, kt << 5);
      v16h bf = load_B_frag(WT, Kin, kt << 5, nt << 4);
      c = __builtin_amdgcn_wmma_f32_16x16x32_f16(false, a, false, bf, (short)0, c, false, false);
    }
    store_C_relu(c, out, mt << 4, Nc, nt << 4, bias);
  }
}

// Final Linear+ReLU layer fused with masked max over groups of 64 rows.
// snv[g] = number of valid rows in group g; invalid rows masked to NEGF.
// ATOMIC=false: out row per group at outBase + g*ldOut. ATOMIC=true: atomicMax per col.
template <bool ATOMIC>
__device__ inline void mlp_final(const _Float16* __restrict__ in,
                                 const _Float16* __restrict__ WT, const float* __restrict__ bias,
                                 float* __restrict__ outBase, int ldOut,
                                 int Kin, int Nc, const int* snv, int ngroups, int validGroups,
                                 int wave, int nwaves) {
  int NT = Nc >> 4, KT = Kin >> 5;
  int lane = threadIdx.x & 31;
  int hi = lane >> 4, nlo = lane & 15;
  for (int tix = wave; tix < NT * ngroups; tix += nwaves) {
    int nt = tix / ngroups, g = tix - nt * ngroups;
    int nv = snv[g];
    float bcol = bias[(nt << 4) + nlo];
    float gmax = NEGF;
    for (int sub = 0; sub < 4; ++sub) {      // 4 M-tiles of 16 rows = 64-row group
      int mt = (g << 2) + sub;
      v8f c = {};
      for (int kt = 0; kt < KT; ++kt) {
        v16h a  = load_A_frag(in, mt << 4, Kin, kt << 5);
        v16h bf = load_B_frag(WT, Kin, kt << 5, nt << 4);
        c = __builtin_amdgcn_wmma_f32_16x16x32_f16(false, a, false, bf, (short)0, c, false, false);
      }
#pragma unroll
      for (int r = 0; r < 8; ++r) {
        int j = (sub << 4) + r + (hi << 3);  // row index within the 64-row group
        float v = c[r] + bcol;
        v = v > 0.f ? v : 0.f;
        gmax = fmaxf(gmax, j < nv ? v : NEGF);
      }
    }
    gmax = fmaxf(gmax, __shfl_xor(gmax, 16, 32));  // combine M0-7 / M8-15 lane halves
    if (lane < 16 && g < validGroups) {
      int col = (nt << 4) + nlo;
      if (ATOMIC) {
        atomicMax((int*)(outBase + col), __float_as_int(gmax)); // vals >= 0 -> monotone
      } else {
        outBase[(size_t)g * ldOut + col] = gmax;
      }
    }
  }
}

// ---------------------------------------------------------------------------
// Support kernels
// ---------------------------------------------------------------------------

// f32 W[Kin][Nc] -> f16 transposed, K-padded WT[Nc][Kpad]
__global__ void convert_w_kernel(const float* __restrict__ W, _Float16* __restrict__ dst,
                                 int Kin, int Kpad, int Nc) {
  int i = blockIdx.x * 256 + threadIdx.x;
  int tot = Kpad * Nc;
  if (i >= tot) return;
  int n = i / Kpad, k = i - n * Kpad;   // contiguous writes along K of WT
  dst[i] = (k < Kin) ? (_Float16)W[(size_t)k * Nc + n] : (_Float16)0.f;
}

__global__ void zero_kernel(float* p, int n) {
  int i = blockIdx.x * 256 + threadIdx.x;
  if (i < n) p[i] = 0.f;
}

// Farthest point sampling: one workgroup per cloud. Positions + min-dist in LDS.
__global__ void __launch_bounds__(256) fps_kernel(const float* __restrict__ pos,
                                                  int N, int M, int* __restrict__ cidx) {
  __shared__ float sx[1024], sy[1024], sz[1024], md[1024];
  __shared__ float wval[8];
  __shared__ int   widx[8];
  __shared__ int   curIdx;
  int b = blockIdx.x;
  int t = threadIdx.x;
  for (int i = t; i < 1024; i += 256) {
    if (i < N) {
      const float* p = pos + ((size_t)b * N + i) * 3;
      sx[i] = p[0]; sy[i] = p[1]; sz[i] = p[2];
    } else { sx[i] = 0.f; sy[i] = 0.f; sz[i] = 0.f; }
  }
  __syncthreads();
  float x0 = sx[0], y0 = sy[0], z0 = sz[0];
  for (int i = t; i < 1024; i += 256) {
    if (i < N) {
      float dx = sx[i] - x0, dy = sy[i] - y0, dz = sz[i] - z0;
      md[i] = dx * dx + dy * dy + dz * dz;
    } else md[i] = -1.0e30f;
  }
  if (t == 0) cidx[(size_t)b * M] = 0;
  __syncthreads();
  for (int s = 1; s < M; ++s) {
    // argmax(md), first index on ties (matches jnp.argmax)
    float bv = -2.0e30f; int bi = 0;
    for (int i = t; i < 1024; i += 256) {
      float v = md[i];
      if (v > bv) { bv = v; bi = i; }  // ascending i keeps first max per thread
    }
    for (int off = 16; off > 0; off >>= 1) {
      float ov = __shfl_down(bv, off, 32);
      int   oi = __shfl_down(bi, off, 32);
      if (ov > bv || (ov == bv && oi < bi)) { bv = ov; bi = oi; }
    }
    if ((t & 31) == 0) { wval[t >> 5] = bv; widx[t >> 5] = bi; }
    __syncthreads();
    if (t == 0) {
      float fv = wval[0]; int fi = widx[0];
      for (int w = 1; w < 8; ++w)
        if (wval[w] > fv || (wval[w] == fv && widx[w] < fi)) { fv = wval[w]; fi = widx[w]; }
      curIdx = fi;
      cidx[(size_t)b * M + s] = fi;
    }
    __syncthreads();
    int ci = curIdx;
    float cx = sx[ci], cy = sy[ci], cz = sz[ci];
    for (int i = t; i < 1024; i += 256) {
      if (i < N) {
        float dx = sx[i] - cx, dy = sy[i] - cy, dz = sz[i] - cz;
        md[i] = fminf(md[i], dx * dx + dy * dy + dz * dz);
      }
    }
    __syncthreads();
  }
}

__global__ void gather_kernel(const float* __restrict__ src, const int* __restrict__ cidx,
                              float* __restrict__ dst, int Nsrc, int M) {
  int i = blockIdx.x * 256 + threadIdx.x;
  if (i >= BB * M) return;
  int b = i / M;
  int idx = cidx[i];
  const float* s = src + ((size_t)b * Nsrc + idx) * 3;
  float* d = dst + (size_t)i * 3;
  d[0] = s[0]; d[1] = s[1]; d[2] = s[2];
}

// Ball query: one wave per center, first-K in index order (torch_cluster radius).
// Pads unused slots with the center's own source index (rel=0, masked later).
__global__ void __launch_bounds__(256) ballquery_kernel(
    const float* __restrict__ src, const float* __restrict__ ctr,
    const int* __restrict__ cidx, int N, int M, float r2,
    int* __restrict__ nbr, int* __restrict__ nvalid) {
  int b = blockIdx.y;
  int m = blockIdx.x * 8 + (threadIdx.x >> 5);
  if (m >= M) return;
  int lane = threadIdx.x & 31;
  const float* cp = ctr + ((size_t)b * M + m) * 3;
  float cx = cp[0], cy = cp[1], cz = cp[2];
  int* list = nbr + ((size_t)b * M + m) * KNN;
  int count = 0;
  for (int base = 0; base < N; base += 32) {
    int i = base + lane;
    bool ok = false;
    if (i < N) {
      const float* p = src + ((size_t)b * N + i) * 3;
      float dx = p[0] - cx, dy = p[1] - cy, dz = p[2] - cz;
      ok = (dx * dx + dy * dy + dz * dz) <= r2;
    }
    unsigned msk = (unsigned)__ballot(ok);          // wave32: low 32 bits
    int pre = __popc(msk & ((1u << lane) - 1u));
    if (ok) {
      int p = count + pre;
      if (p < KNN) list[p] = i;
    }
    count += __popc(msk);
    if (count >= KNN) break;                         // uniform across wave
  }
  int total = count < KNN ? count : KNN;
  int selfIdx = cidx[(size_t)b * M + m];
  for (int p = total + lane; p < KNN; p += 32) list[p] = selfIdx;
  if (lane == 0) nvalid[(size_t)b * M + m] = total;
}

// ---------------------------------------------------------------------------
// Fused SA modules: feat gather -> MLP (WMMA) -> masked max, all in LDS.
// 256 threads = 8 waves; 4 centers x 64 neighbors = 256 rows per block.
// ---------------------------------------------------------------------------

__global__ void __launch_bounds__(256) sa1_kernel(
    const float* __restrict__ pos, const float* __restrict__ ctr,
    const int* __restrict__ nbr, const int* __restrict__ nvalid,
    const _Float16* w0, const _Float16* w1, const _Float16* w2,
    const float* b0, const float* b1, const float* b2,
    float* __restrict__ xout) {
  extern __shared__ _Float16 sm[];
  _Float16* feat = sm;                 // [256][32]
  _Float16* act1 = sm + 256 * 32;      // [256][64]
  _Float16* act2 = act1 + 256 * 64;    // [256][64]
  __shared__ int   snv[4];
  __shared__ float scen[4][3];
  int b = blockIdx.y;
  int cbase = blockIdx.x * 4;
  int t = threadIdx.x;
  if (t < 4) {
    int gc = cbase + t;
    snv[t] = (gc < M1C) ? nvalid[b * M1C + gc] : 0;
  }
  if (t < 12) {
    int gc = cbase + t / 3;
    scen[t / 3][t % 3] = (gc < M1C) ? ctr[((size_t)b * M1C + gc) * 3 + t % 3] : 0.f;
  }
  __syncthreads();
  int c = t >> 6, j = t & 63;
  int gc = cbase + c;
  float rx = 0.f, ry = 0.f, rz = 0.f;
  if (gc < M1C) {
    int idx = nbr[((size_t)(b * M1C + gc)) * KNN + j];
    const float* pp = pos + ((size_t)b * N1 + idx) * 3;
    rx = pp[0] - scen[c][0]; ry = pp[1] - scen[c][1]; rz = pp[2] - scen[c][2];
  }
  _Float16* fr = feat + t * 32;
  fr[0] = (_Float16)rx; fr[1] = (_Float16)ry; fr[2] = (_Float16)rz;
  for (int k = 3; k < 32; ++k) fr[k] = (_Float16)0.f;
  __syncthreads();
  int wave = t >> 5;
  mlp_layer(feat, act1, w0, b0, 256, 32, 64, wave, 8);
  __syncthreads();
  mlp_layer(act1, act2, w1, b1, 256, 64, 64, wave, 8);
  __syncthreads();
  int vg = M1C - cbase; vg = vg > 4 ? 4 : vg;
  mlp_final<false>(act2, w2, b2, xout + ((size_t)b * M1C + cbase) * 128, 128,
                   64, 128, snv, 4, vg, wave, 8);
}

__global__ void __launch_bounds__(256) sa2_kernel(
    const float* __restrict__ pos1, const float* __restrict__ x1,
    const float* __restrict__ ctr2,
    const int* __restrict__ nbr, const int* __restrict__ nvalid,
    const _Float16* w0, const _Float16* w1, const _Float16* w2,
    const float* b0, const float* b1, const float* b2,
    float* __restrict__ xout) {
  extern __shared__ _Float16 sm[];
  _Float16* bufA = sm;                 // feat [256][160], later act2 [256][128]
  _Float16* bufB = sm + 256 * 160;     // act1 [256][128]
  __shared__ int   snv[4];
  __shared__ float scen[4][3];
  int b = blockIdx.y;
  int cbase = blockIdx.x * 4;
  int t = threadIdx.x;
  if (t < 4) {
    int gc = cbase + t;
    snv[t] = (gc < M2C) ? nvalid[b * M2C + gc] : 0;
  }
  if (t < 12) {
    int gc = cbase + t / 3;
    scen[t / 3][t % 3] = (gc < M2C) ? ctr2[((size_t)b * M2C + gc) * 3 + t % 3] : 0.f;
  }
  __syncthreads();
  int c = t >> 6, j = t & 63;
  int gc = cbase + c;
  _Float16* fr = bufA + (size_t)t * 160;
  if (gc < M2C) {
    int idx = nbr[((size_t)(b * M2C + gc)) * KNN + j];
    const float* xr = x1 + ((size_t)b * M1C + idx) * 128;
    for (int k = 0; k < 128; ++k) fr[k] = (_Float16)xr[k];
    const float* pp = pos1 + ((size_t)b * M1C + idx) * 3;
    fr[128] = (_Float16)(pp[0] - scen[c][0]);
    fr[129] = (_Float16)(pp[1] - scen[c][1]);
    fr[130] = (_Float16)(pp[2] - scen[c][2]);
  } else {
    for (int k = 0; k < 131; ++k) fr[k] = (_Float16)0.f;
  }
  for (int k = 131; k < 160; ++k) fr[k] = (_Float16)0.f;
  __syncthreads();
  int wave = t >> 5;
  mlp_layer(bufA, bufB, w0, b0, 256, 160, 128, wave, 8);
  __syncthreads();
  mlp_layer(bufB, bufA, w1, b1, 256, 128, 128, wave, 8);  // bufA reused (stride 128)
  __syncthreads();
  int vg = M2C - cbase; vg = vg > 4 ? 4 : vg;
  mlp_final<false>(bufA, w2, b2, xout + ((size_t)b * M2C + cbase) * 256, 256,
                   128, 256, snv, 4, vg, wave, 8);
}

// SA3: 64 centers per block, 5-layer MLP, masked max over valid rows + global
// atomic max into d_out (all values >= 0 post-ReLU, output pre-zeroed).
__global__ void __launch_bounds__(256) sa3_kernel(
    const float* __restrict__ x2, const float* __restrict__ pos2,
    const _Float16* w0, const _Float16* w1, const _Float16* w2,
    const _Float16* w3, const _Float16* w4,
    const float* b0, const float* b1, const float* b2,
    const float* b3, const float* b4,
    float* __restrict__ outp) {
  extern __shared__ _Float16 sm[];
  _Float16* A  = sm;             // [64][288], reused at strides 256
  _Float16* Bf = sm + 64 * 288;  // [64][512], reused at stride 256
  __shared__ int snv[1];
  int b = blockIdx.y;
  int mbase = blockIdx.x * 64;
  int t = threadIdx.x;
  if (t == 0) {
    int v = M2C - mbase;
    v = v < 0 ? 0 : (v > 64 ? 64 : v);
    snv[0] = v;
  }
  int r = t >> 2, q = t & 3;     // 4 threads per row, 72 cols each
  int gm = mbase + r;
  _Float16* fr = A + (size_t)r * 288;
  for (int k = q * 72; k < q * 72 + 72; ++k) {
    float v = 0.f;
    if (gm < M2C) {
      if (k < 256)      v = x2[((size_t)b * M2C + gm) * 256 + k];
      else if (k < 259) v = pos2[((size_t)b * M2C + gm) * 3 + (k - 256)];
    }
    fr[k] = (_Float16)v;
  }
  __syncthreads();
  int wave = t >> 5;
  mlp_layer(A, Bf, w0, b0, 64, 288, 256, wave, 8); __syncthreads();
  mlp_layer(Bf, A, w1, b1, 64, 256, 256, wave, 8); __syncthreads();
  mlp_layer(A, Bf, w2, b2, 64, 256, 512, wave, 8); __syncthreads();
  mlp_layer(Bf, A, w3, b3, 64, 512, 256, wave, 8); __syncthreads();
  mlp_final<true>(A, w4, b4, outp + (size_t)b * 128, 0, 256, 128, snv, 1, 1, wave, 8);
}

// ---------------------------------------------------------------------------
// Host launcher
// ---------------------------------------------------------------------------

extern "C" void kernel_launch(void* const* d_in, const int* in_sizes, int n_in,
                              void* d_out, int out_size, void* d_ws, size_t ws_size,
                              hipStream_t stream) {
  (void)in_sizes; (void)n_in; (void)ws_size;
  const float* joints = (const float*)d_in[0];

  // {W idx, b idx, Kin, Kpad, Nc} per layer, in d_in pytree order
  static const int wcfg[11][5] = {
      {2, 3, 3, 32, 64},     {4, 5, 64, 64, 64},    {6, 7, 64, 64, 128},
      {8, 9, 131, 160, 128}, {10, 11, 128, 128, 128}, {12, 13, 128, 128, 256},
      {14, 15, 259, 288, 256}, {16, 17, 256, 256, 256}, {18, 19, 256, 256, 512},
      {20, 21, 512, 512, 256}, {22, 23, 256, 256, 128}};
  size_t whalf[12];
  whalf[0] = 0;
  for (int i = 0; i < 11; ++i)
    whalf[i + 1] = whalf[i] + (size_t)wcfg[i][3] * wcfg[i][4];

  char* ws = (char*)d_ws;
  size_t off = 0;
  auto take = [&](size_t bytes) {
    void* p = ws + off;
    off = (off + bytes + 255) & ~(size_t)255;
    return p;
  };
  _Float16* wgt  = (_Float16*)take(whalf[11] * sizeof(_Float16));
  int*   cidx1   = (int*)take((size_t)BB * M1C * 4);
  float* pos1    = (float*)take((size_t)BB * M1C * 3 * 4);
  int*   nbr1    = (int*)take((size_t)BB * M1C * KNN * 4);
  int*   nvalid1 = (int*)take((size_t)BB * M1C * 4);
  float* x1      = (float*)take((size_t)BB * M1C * 128 * 4);
  int*   cidx2   = (int*)take((size_t)BB * M2C * 4);
  float* pos2    = (float*)take((size_t)BB * M2C * 3 * 4);
  int*   nbr2    = (int*)take((size_t)BB * M2C * KNN * 4);
  int*   nvalid2 = (int*)take((size_t)BB * M2C * 4);
  float* x2      = (float*)take((size_t)BB * M2C * 256 * 4);

  // 1. Weights -> f16, transposed to [Nc][Kpad], K zero-padded to mult of 32.
  for (int i = 0; i < 11; ++i) {
    int tot = wcfg[i][3] * wcfg[i][4];
    convert_w_kernel<<<(tot + 255) / 256, 256, 0, stream>>>(
        (const float*)d_in[wcfg[i][0]], wgt + whalf[i], wcfg[i][2], wcfg[i][3], wcfg[i][4]);
  }
  // 2-4. Stage 1 sampling/grouping
  fps_kernel<<<BB, 256, 0, stream>>>(joints, N1, M1C, cidx1);
  gather_kernel<<<(BB * M1C + 255) / 256, 256, 0, stream>>>(joints, cidx1, pos1, N1, M1C);
  ballquery_kernel<<<dim3((M1C + 7) / 8, BB), 256, 0, stream>>>(
      joints, pos1, cidx1, N1, M1C, 0.4f * 0.4f, nbr1, nvalid1);
  // 5. SA1 fused MLP (LDS: 80 KB)
  sa1_kernel<<<dim3((M1C + 3) / 4, BB), 256, 81920, stream>>>(
      joints, pos1, nbr1, nvalid1,
      wgt + whalf[0], wgt + whalf[1], wgt + whalf[2],
      (const float*)d_in[3], (const float*)d_in[5], (const float*)d_in[7], x1);
  // 6-8. Stage 2 sampling/grouping
  fps_kernel<<<BB, 256, 0, stream>>>(pos1, M1C, M2C, cidx2);
  gather_kernel<<<(BB * M2C + 255) / 256, 256, 0, stream>>>(pos1, cidx2, pos2, M1C, M2C);
  ballquery_kernel<<<dim3((M2C + 7) / 8, BB), 256, 0, stream>>>(
      pos1, pos2, cidx2, M1C, M2C, 0.6f * 0.6f, nbr2, nvalid2);
  // 9. SA2 fused MLP (LDS: 144 KB)
  sa2_kernel<<<dim3((M2C + 3) / 4, BB), 256, 147456, stream>>>(
      pos1, x1, pos2, nbr2, nvalid2,
      wgt + whalf[3], wgt + whalf[4], wgt + whalf[5],
      (const float*)d_in[9], (const float*)d_in[11], (const float*)d_in[13], x2);
  // 10-11. SA3 + global max pool
  zero_kernel<<<(out_size + 255) / 256, 256, 0, stream>>>((float*)d_out, out_size);
  sa3_kernel<<<dim3((M2C + 63) / 64, BB), 256, 102400, stream>>>(
      x2, pos2,
      wgt + whalf[6], wgt + whalf[7], wgt + whalf[8], wgt + whalf[9], wgt + whalf[10],
      (const float*)d_in[15], (const float*)d_in[17], (const float*)d_in[19],
      (const float*)d_in[21], (const float*)d_in[23], (float*)d_out);
}
